// NonSeparable_22677427323146
// MI455X (gfx1250) — matrix operand order repaired and checked by reading, not compiled, not run on hardware
//
#include <hip/hip_runtime.h>

// ---------- types ----------
typedef __attribute__((ext_vector_type(16))) __bf16 v16bf;
typedef __attribute__((ext_vector_type(8)))  float  v8f;
typedef __attribute__((ext_vector_type(4)))  float  v4f;
typedef __attribute__((ext_vector_type(4)))  unsigned int v4u;

struct KPtrs { const float* p[21]; };

#define NUM_LVL   21
#define H_IMG     512
#define W_IMG     512
#define SUM_S     483              // sum of (2i+3), i=0..20
#define FRAGS     (SUM_S * 3 * 2)  // (cum+ky) * channel * chunk = 2898
#define FRAG_ELEM 512              // 32 lanes * 16 bf16
#define LDS_ROWS  58
#define LDS_PITCH 72               // elements; 144B rows -> 16B aligned, bank stride gcd 4
#define WBUF_OFF_BYTES (176 * 1024)

__device__ __forceinline__ unsigned short f2bf(float f) {
  unsigned int u = __float_as_uint(f);
  unsigned int r = u + 0x7FFFu + ((u >> 16) & 1u);   // round-to-nearest-even
  return (unsigned short)(r >> 16);
}

__device__ __forceinline__ v16bf load16bf(const unsigned short* pt) {
  union { v4u u[2]; v16bf v; } t;
  t.u[0] = *(const v4u*)pt;
  t.u[1] = *((const v4u*)pt + 1);
  return t.v;
}

// ---------------------------------------------------------------------------
// Kernel 1: symmetrize (kp + kp^T), reflect-pad to (2p+1)^2, softmax -> f32
// grid = 63 (level*3 + channel), block = 256
// ---------------------------------------------------------------------------
__global__ __launch_bounds__(256) void prep_weights(KPtrs ks, float* wbuf) {
  const int lvl = blockIdx.x / 3, c = blockIdx.x % 3;
  const int q = lvl + 2, p = lvl + 1, s = 2 * p + 1;
  const float* kp = ks.p[lvl] + c * q * q;

  __shared__ float sym[23 * 23];
  __shared__ float red[256];
  __shared__ float sstat[2];
  const int tid = threadIdx.x;

  for (int idx = tid; idx < q * q; idx += 256) {
    int h = idx / q, w = idx % q;
    sym[idx] = kp[h * q + w] + kp[w * q + h];  // kp + rot/flip == kp + kp^T
  }
  __syncthreads();

  // max over padded s*s (reflect: no edge repeat)
  float mx = -1e30f;
  for (int idx = tid; idx < s * s; idx += 256) {
    int h = idx / s, w = idx % s;
    int rh = h < q ? h : 2 * q - 2 - h;
    int rw = w < q ? w : 2 * q - 2 - w;
    mx = fmaxf(mx, sym[rh * q + rw]);
  }
  red[tid] = mx; __syncthreads();
  for (int st = 128; st > 0; st >>= 1) {
    if (tid < st) red[tid] = fmaxf(red[tid], red[tid + st]);
    __syncthreads();
  }
  if (tid == 0) sstat[0] = red[0];
  __syncthreads();
  mx = sstat[0];

  float sum = 0.0f;
  for (int idx = tid; idx < s * s; idx += 256) {
    int h = idx / s, w = idx % s;
    int rh = h < q ? h : 2 * q - 2 - h;
    int rw = w < q ? w : 2 * q - 2 - w;
    sum += __expf(sym[rh * q + rw] - mx);
  }
  red[tid] = sum; __syncthreads();
  for (int st = 128; st > 0; st >>= 1) {
    if (tid < st) red[tid] += red[tid + st];
    __syncthreads();
  }
  if (tid == 0) sstat[1] = red[0];
  __syncthreads();
  const float inv = 1.0f / sstat[1];

  int wbase = 0;
  for (int t = 0; t < lvl; ++t) { int ss = 2 * t + 3; wbase += 3 * ss * ss; }
  float* wout = wbuf + wbase + c * s * s;
  for (int idx = tid; idx < s * s; idx += 256) {
    int h = idx / s, w = idx % s;
    int rh = h < q ? h : 2 * q - 2 - h;
    int rw = w < q ? w : 2 * q - 2 - w;
    wout[idx] = __expf(sym[rh * q + rw] - mx) * inv;
  }
}

// ---------------------------------------------------------------------------
// Kernel 2: build bf16 A-fragments (banded Toeplitz in WMMA A layout).
// fi = ((cum(i)+ky)*3 + c)*2 + chunk ; one wave (32 lanes) per fragment.
// Lane L holds row m=L%16; element e -> K = e + 16*half, half=L/16.
// A[m][j] = w[ky][j - m - (24-p)], j = 32*chunk + K.
// ---------------------------------------------------------------------------
__global__ __launch_bounds__(32) void prep_frags(const float* wbuf,
                                                 unsigned short* frag) {
  const int fi = blockIdx.x;
  const int chunk = fi & 1;
  const int t = fi >> 1;
  const int c = t % 3;
  const int r = t / 3;                    // cum(i) + ky, in [0, 483)
  int lvl = 0;
  while ((lvl + 1) * (lvl + 3) <= r) ++lvl;   // cum(i) = i*(i+2)
  const int ky = r - lvl * (lvl + 2);
  const int p = lvl + 1, s = 2 * p + 1;

  int wbase = 0;
  for (int tt = 0; tt < lvl; ++tt) { int ss = 2 * tt + 3; wbase += 3 * ss * ss; }
  const float* w = wbuf + wbase + (c * s + ky) * s;

  const int lane = threadIdx.x;
  const int m = lane & 15, half = lane >> 4;
  unsigned short* out = frag + (size_t)fi * FRAG_ELEM + lane * 16;
#pragma unroll
  for (int e = 0; e < 16; ++e) {
    int K = e + 16 * half;
    int j = 32 * chunk + K;
    int kx = j - m - (24 - p);
    float v = (kx >= 0 && kx < s) ? w[kx] : 0.0f;
    out[e] = f2bf(v);
  }
}

// ---------------------------------------------------------------------------
// Kernel 3: conv. grid=(32,32,12) tiles x images; block=(32,22): 21 blur
// waves + 1 copy wave. LDS-staged 58x64 bf16 halo tile (fixed origin
// (y0-21, x0-24)), reflect-indexed. Per wave: loop ky, 2 K-chunks of
// v_wmma_f32_16x16x32_bf16, store f32 D tile.
// ---------------------------------------------------------------------------
__global__ __launch_bounds__(704) void conv_kernel(const float* __restrict__ x,
                                                   const unsigned short* __restrict__ frag,
                                                   float* __restrict__ out) {
  __shared__ alignas(16) unsigned short lds[LDS_ROWS * LDS_PITCH];

  const int x0 = blockIdx.x * 16, y0 = blockIdx.y * 16;
  const int img = blockIdx.z;               // b*3 + c
  const int b = img / 3, cch = img % 3;
  const float* xin = x + (size_t)img * H_IMG * W_IMG;

  const int tid = threadIdx.y * 32 + threadIdx.x;
  for (int idx = tid; idx < LDS_ROWS * 64; idx += 704) {
    int row = idx >> 6, col = idx & 63;
    int gy = y0 - 21 + row; gy = gy < 0 ? -gy : (gy > 511 ? 1022 - gy : gy);
    int gx = x0 - 24 + col; gx = gx < 0 ? -gx : (gx > 511 ? 1022 - gx : gx);
    lds[row * LDS_PITCH + col] = f2bf(xin[gy * W_IMG + gx]);
  }
  __syncthreads();

  const int wv = threadIdx.y;
  const int lane = threadIdx.x;
  const int n = lane & 15, half = lane >> 4;

  if (wv == NUM_LVL) {                      // pass-through: out[:,0:3] = x
    for (int idx = lane; idx < 256; idx += 32) {
      int ly = idx >> 4, lx = idx & 15;
      out[(((size_t)b * 66 + cch) * H_IMG + (y0 + ly)) * W_IMG + (x0 + lx)] =
          xin[(y0 + ly) * W_IMG + (x0 + lx)];
    }
    return;
  }

  const int lvl = wv, p = lvl + 1, s = 2 * p + 1;
  v8f acc = {0.f, 0.f, 0.f, 0.f, 0.f, 0.f, 0.f, 0.f};

  // fragment byte layout: fi = (cum(i)+ky)*6 + c*2 + chunk, FRAG_ELEM shorts each
  const unsigned short* fp0 =
      frag + (size_t)(lvl * (lvl + 2) * 6 + cch * 2) * FRAG_ELEM + lane * 16;

  for (int ky = 0; ky < s; ++ky) {
    const int r0 = ky - p + 21;             // base LDS row (lane adds n)
    const unsigned short* fky = fp0 + (size_t)ky * 6 * FRAG_ELEM;
    const unsigned short* brow = &lds[(r0 + n) * LDS_PITCH + 16 * half];
#pragma unroll
    for (int ch = 0; ch < 2; ++ch) {
      v16bf a = load16bf(fky + ch * FRAG_ELEM);       // 2x global_load_b128
      v16bf bb = load16bf(brow + 32 * ch);            // 2x ds_load_b128 (16B aligned)
      acc = __builtin_amdgcn_wmma_f32_16x16x32_bf16(
          false, a, false, bb, (short)0, acc, false, false);
    }
  }

  // D layout: lane -> n = L%16, cols m = j + 8*half ; contiguous 8 floats
  float* op = out + (((size_t)b * 66 + 3 + 3 * lvl + cch) * H_IMG + (y0 + n)) * W_IMG
                  + x0 + 8 * half;
  v4f lo = __builtin_shufflevector(acc, acc, 0, 1, 2, 3);
  v4f hi = __builtin_shufflevector(acc, acc, 4, 5, 6, 7);
  *(v4f*)op = lo;
  *((v4f*)op + 1) = hi;
}

// ---------------------------------------------------------------------------
extern "C" void kernel_launch(void* const* d_in, const int* in_sizes, int n_in,
                              void* d_out, int out_size, void* d_ws, size_t ws_size,
                              hipStream_t stream) {
  (void)in_sizes; (void)n_in; (void)out_size; (void)ws_size;
  const float* x = (const float*)d_in[0];
  KPtrs ks;
  for (int i = 0; i < NUM_LVL; ++i) ks.p[i] = (const float*)d_in[1 + i];

  float* wbuf = (float*)d_ws;                                   // 3*14189 f32 = 166KB
  unsigned short* frag = (unsigned short*)((char*)d_ws + WBUF_OFF_BYTES); // ~2.9MB
  float* out = (float*)d_out;

  prep_weights<<<63, 256, 0, stream>>>(ks, wbuf);
  prep_frags<<<FRAGS, 32, 0, stream>>>(wbuf, frag);

  dim3 grid(W_IMG / 16, H_IMG / 16, 12);
  dim3 blk(32, NUM_LVL + 1, 1);
  conv_kernel<<<grid, blk, 0, stream>>>(x, frag, out);
}
